// AlphaZeroGNN_56805237457555
// MI455X (gfx1250) — matrix operand order repaired and checked by reading, not compiled, not run on hardware
//
#include <hip/hip_runtime.h>

// ---------------------------------------------------------------------------
// Types for CDNA5 WMMA (wave32): v_wmma_f32_16x16x32_bf16
// ---------------------------------------------------------------------------
typedef float  v8f   __attribute__((ext_vector_type(8)));
typedef __bf16 v16bf __attribute__((ext_vector_type(16)));
typedef __bf16 v8bf  __attribute__((ext_vector_type(8)));
typedef __bf16 v4bf  __attribute__((ext_vector_type(4)));

#define N_NODES 32768
#define N_EDGES 262144
#define N_B     128
#define N_M     512
#define HID     256

// ---------------------------------------------------------------------------
// CDNA5 async global -> LDS (ASYNCcnt), 16 bytes per op, via inline asm.
// LDS byte address = low 32 bits of the generic shared-memory pointer
// (ISA 10.2: LDS aperture).
// ---------------------------------------------------------------------------
__device__ __forceinline__ void async_ld16(const void* gptr, void* lptr) {
    unsigned loff = (unsigned)(unsigned long long)lptr;
    asm volatile("global_load_async_to_lds_b128 %0, %1, off"
                 :: "v"(loff), "v"(gptr) : "memory");
}

__device__ __forceinline__ void wait_async0() {
#if __has_builtin(__builtin_amdgcn_s_wait_asynccnt)
    __builtin_amdgcn_s_wait_asynccnt(0);
#else
    asm volatile("s_wait_asynccnt 0x0" ::: "memory");
#endif
}

// ---------------------------------------------------------------------------
// Weight convert + transpose: W[K][Nn] f32  ->  WT[Nn][K] bf16
// ---------------------------------------------------------------------------
__global__ void __launch_bounds__(256) wtrans_kernel(const float* __restrict__ W,
                                                     __bf16* __restrict__ WT,
                                                     int K, int Nn) {
    int i = blockIdx.x * 256 + threadIdx.x;
    if (i < K * Nn) {
        int k = i / Nn, n = i % Nn;
        WT[(long)n * K + k] = (__bf16)W[i];
    }
}

// f32 -> bf16, 4 elements/thread
__global__ void __launch_bounds__(256) cvt_bf16_kernel(const float* __restrict__ in,
                                                       __bf16* __restrict__ out, long n4) {
    long i = (long)blockIdx.x * 256 + threadIdx.x;
    if (i < n4) {
        float4 v = ((const float4*)in)[i];
        v4bf o;
        o[0] = (__bf16)v.x; o[1] = (__bf16)v.y; o[2] = (__bf16)v.z; o[3] = (__bf16)v.w;
        *(v4bf*)&out[i * 4] = o;
    }
}

__global__ void __launch_bounds__(256) copy4_kernel(const float* __restrict__ in,
                                                    float* __restrict__ out, long n4) {
    long i = (long)blockIdx.x * 256 + threadIdx.x;
    if (i < n4) ((float4*)out)[i] = ((const float4*)in)[i];
}

// ---------------------------------------------------------------------------
// Input projection: h0 = [x | onehot(player[batch])] @ Win + b_in   (K=34)
// ---------------------------------------------------------------------------
__global__ void __launch_bounds__(256) input_proj_kernel(const float* __restrict__ x,
                                                         const int* __restrict__ player,
                                                         const int* __restrict__ batch,
                                                         const float* __restrict__ Win,
                                                         const float* __restrict__ b_in,
                                                         float* __restrict__ h0) {
    long i = (long)blockIdx.x * 256 + threadIdx.x;    // N*H threads
    int row = (int)(i >> 8), col = (int)(i & 255);
    const float* xr = x + (long)row * 32;
    float s = b_in[col];
#pragma unroll
    for (int k = 0; k < 32; ++k) s = fmaf(xr[k], Win[k * HID + col], s);
    int p = player[batch[row]];
    s += Win[(32 + p) * HID + col];
    h0[i] = s;
}

// ---------------------------------------------------------------------------
// Edge scatter: z[dst] += h[src]   (thread = one edge x 4-float chunk)
// ---------------------------------------------------------------------------
__global__ void __launch_bounds__(256) scatter_kernel(const int* __restrict__ ei,
                                                      const float* __restrict__ h,
                                                      float* __restrict__ z) {
    long tid = (long)blockIdx.x * 256 + threadIdx.x;  // E * 64
    long e = tid >> 6;
    int c = (int)(tid & 63) * 4;
    int src = ei[e], dst = ei[N_EDGES + e];
    float4 v = *(const float4*)&h[(long)src * HID + c];
    float* zp = &z[(long)dst * HID + c];
    atomicAdd(zp + 0, v.x); atomicAdd(zp + 1, v.y);
    atomicAdd(zp + 2, v.z); atomicAdd(zp + 3, v.w);
}

// ---------------------------------------------------------------------------
// Shared-memory layout for WMMA GEMM kernels:
// async staging tiles overlaid with the f32 epilogue tile.
// ---------------------------------------------------------------------------
union __align__(64) GemmSmem {
    struct { __bf16 As[64][32]; __bf16 Bs[256][32]; } t;  // 4 + 16 KB
    float c[64][256];                                     // 64 KB
};

__device__ __forceinline__ v16bf load_afrag(const __bf16 (*As)[32], int rt, int lane) {
    // A frag, k order used here (matches B): lane half hi=lane>>4 holds 16 consecutive K = hi*16..+16
    int m = lane & 15, hi = lane >> 4;
    v8bf lo = *(const v8bf*)&As[rt * 16 + m][hi * 16];
    v8bf hh = *(const v8bf*)&As[rt * 16 + m][hi * 16 + 8];
    v16bf a;
#pragma unroll
    for (int i = 0; i < 8; ++i) { a[i] = lo[i]; a[8 + i] = hh[i]; }
    return a;
}

// stage one 64x32 A tile + 256x32 B tile via async DMA (5 x b128 per thread)
__device__ __forceinline__ void stage_tiles(GemmSmem& sm,
                                            const __bf16* __restrict__ Arow0,
                                            int k0, int ldw,
                                            const __bf16* __restrict__ WT, int t) {
    int r = t >> 2, kc = (t & 3) * 8;
    async_ld16(Arow0 + (long)r * 256 + k0 + kc, &sm.t.As[r][kc]);
    const __bf16* src = &WT[(long)t * ldw + k0];
    async_ld16(src + 0,  &sm.t.Bs[t][0]);
    async_ld16(src + 8,  &sm.t.Bs[t][8]);
    async_ld16(src + 16, &sm.t.Bs[t][16]);
    async_ld16(src + 24, &sm.t.Bs[t][24]);
}

// ---------------------------------------------------------------------------
// WMMA GEMM + bias + layernorm (+ optional residual) + relu
//   out[r, :] = act( LN( A[r,:] @ W + bias ) (+ resid) ),  K = N = 256
// Block: 256 threads (8 waves), tile 64 rows x 256 cols.
// Wave w: row-subtile rt=w&3 (16 rows), col half ch=w>>2 (8 x 16-col subtiles).
// ---------------------------------------------------------------------------
__global__ void __launch_bounds__(256) gemm_ln_kernel(const __bf16* __restrict__ A,
                                                      const __bf16* __restrict__ WT,
                                                      const float* __restrict__ bias,
                                                      const float* __restrict__ lng,
                                                      const float* __restrict__ lnb,
                                                      const float* __restrict__ resid,
                                                      float* __restrict__ out,
                                                      int useResid) {
    __shared__ GemmSmem sm;
    const int t = threadIdx.x, lane = t & 31, wid = t >> 5;
    const int rt = wid & 3, ch = wid >> 2;
    const int hi = lane >> 4, nn = lane & 15;
    const long rowBase = (long)blockIdx.x * 64;
    const __bf16* Arow0 = A + rowBase * 256;

    v8f acc[8];
#pragma unroll
    for (int j = 0; j < 8; ++j)
#pragma unroll
        for (int i = 0; i < 8; ++i) acc[j][i] = 0.f;

    for (int k0 = 0; k0 < 256; k0 += 32) {
        stage_tiles(sm, Arow0, k0, 256, WT, t);
        wait_async0();
        __syncthreads();

        v16bf a = load_afrag(sm.t.As, rt, lane);
#pragma unroll
        for (int j = 0; j < 8; ++j) {
            int n0 = ch * 128 + j * 16;
            // B frag, k order used here (matches A): lane half holds 16 consecutive K for col n
            v16bf b = *(const v16bf*)&sm.t.Bs[n0 + nn][hi * 16];
            acc[j] = __builtin_amdgcn_wmma_f32_16x16x32_bf16(
                false, a, false, b, (short)0, acc[j], false, false);
        }
        __syncthreads();
    }

    // spill D (f32 16x16 layout: VGPR r -> M = hi*8 + r, N = lane&15) + bias
#pragma unroll
    for (int j = 0; j < 8; ++j) {
        int col = ch * 128 + j * 16 + nn;
        float bcol = bias[col];
#pragma unroll
        for (int r = 0; r < 8; ++r) sm.c[rt * 16 + hi * 8 + r][col] = acc[j][r] + bcol;
    }
    __syncthreads();

    { // layernorm: 4 lanes per row
        int row = t >> 2, seg = t & 3;
        float s = 0.f, s2 = 0.f;
        for (int i = 0; i < 64; ++i) {
            float v = sm.c[row][seg * 64 + i];
            s += v; s2 += v * v;
        }
        s  += __shfl_xor(s, 1, 4);  s  += __shfl_xor(s, 2, 4);
        s2 += __shfl_xor(s2, 1, 4); s2 += __shfl_xor(s2, 2, 4);
        float mean = s * (1.f / 256.f);
        float var  = s2 * (1.f / 256.f) - mean * mean;
        float rstd = rsqrtf(var + 1e-5f);
        long grow = (rowBase + row) * 256;
        for (int i = 0; i < 64; ++i) {
            int c = seg * 64 + i;
            float v = (sm.c[row][c] - mean) * rstd * lng[c] + lnb[c];
            if (useResid) v += resid[grow + c];
            out[grow + c] = v > 0.f ? v : 0.f;
        }
    }
}

// ---------------------------------------------------------------------------
// Policy head: per (b,m) row  logit = relu([h[s]|h[t]] @ Wp1 + bp1) . Wp2 + bp2
// Fused gather (async A stage) + WMMA GEMM (K=512) + relu-dot epilogue + mask.
// ---------------------------------------------------------------------------
__global__ void __launch_bounds__(256) policy_kernel(const __bf16* __restrict__ Hb,
                                                     const __bf16* __restrict__ WTp,
                                                     const float* __restrict__ bp1,
                                                     const float* __restrict__ Wp2,
                                                     const float* __restrict__ bp2,
                                                     const int* __restrict__ move,
                                                     float* __restrict__ pol) {
    __shared__ GemmSmem sm;
    __shared__ int srow[64], trow[64], vmask[64];
    const int t = threadIdx.x, lane = t & 31, wid = t >> 5;
    const int rt = wid & 3, ch = wid >> 2;
    const int hi = lane >> 4, nn = lane & 15;
    const long rowBase = (long)blockIdx.x * 64;

    if (t < 64) {
        long gid = rowBase + t;
        int m0 = move[gid * 2], m1 = move[gid * 2 + 1];
        srow[t] = m0 < 0 ? 0 : m0;
        trow[t] = m1 < 0 ? 0 : m1;
        vmask[t] = (m0 != -1 && m1 != -1) ? 1 : 0;
    }
    __syncthreads();

    v8f acc[8];
#pragma unroll
    for (int j = 0; j < 8; ++j)
#pragma unroll
        for (int i = 0; i < 8; ++i) acc[j][i] = 0.f;

    const int r4 = t >> 2, kc4 = (t & 3) * 8;

    for (int it = 0; it < 16; ++it) {
        { // fused gather A stage: cols <256 from h[s], >=256 from h[t]
            int k = it * 32 + kc4;
            int sr = (k < 256) ? srow[r4] : trow[r4];
            async_ld16(&Hb[(long)sr * 256 + (k & 255)], &sm.t.As[r4][kc4]);
            const __bf16* src = &WTp[(long)t * 512 + it * 32];
            async_ld16(src + 0,  &sm.t.Bs[t][0]);
            async_ld16(src + 8,  &sm.t.Bs[t][8]);
            async_ld16(src + 16, &sm.t.Bs[t][16]);
            async_ld16(src + 24, &sm.t.Bs[t][24]);
        }
        wait_async0();
        __syncthreads();

        v16bf a = load_afrag(sm.t.As, rt, lane);
#pragma unroll
        for (int j = 0; j < 8; ++j) {
            int n0 = ch * 128 + j * 16;
            v16bf b = *(const v16bf*)&sm.t.Bs[n0 + nn][hi * 16];
            acc[j] = __builtin_amdgcn_wmma_f32_16x16x32_bf16(
                false, a, false, b, (short)0, acc[j], false, false);
        }
        __syncthreads();
    }

#pragma unroll
    for (int j = 0; j < 8; ++j) {
        int col = ch * 128 + j * 16 + nn;
        float bcol = bp1[col];
#pragma unroll
        for (int r = 0; r < 8; ++r) sm.c[rt * 16 + hi * 8 + r][col] = acc[j][r] + bcol;
    }
    __syncthreads();

    { // epilogue: relu + dot with Wp2, 4 lanes per row, mask invalid moves
        int row = t >> 2, seg = t & 3;
        float d = 0.f;
        for (int i = 0; i < 64; ++i) {
            int c = seg * 64 + i;
            float v = sm.c[row][c];
            v = v > 0.f ? v : 0.f;
            d = fmaf(v, Wp2[c], d);
        }
        d += __shfl_xor(d, 1, 4); d += __shfl_xor(d, 2, 4);
        if (seg == 0) {
            long gid = rowBase + row;
            pol[gid] = vmask[row] ? (d + bp2[0]) : -1e9f;
        }
    }
}

// ---------------------------------------------------------------------------
// Mean pool (graphs are 256 contiguous nodes) + value head
// ---------------------------------------------------------------------------
__global__ void __launch_bounds__(256) meanpool_kernel(const float* __restrict__ h,
                                                       float* __restrict__ ge) {
    int g = blockIdx.x, t = threadIdx.x;
    float s = 0.f;
    for (int r = 0; r < 256; ++r) s += h[((long)g * 256 + r) * 256 + t];
    ge[g * 256 + t] = s * (1.f / 256.f);
}

__global__ void __launch_bounds__(256) dense_bn_relu_kernel(const float* __restrict__ in,
                                                            const float* __restrict__ W,
                                                            const float* __restrict__ b,
                                                            const float* __restrict__ bg,
                                                            const float* __restrict__ bb,
                                                            const float* __restrict__ bm,
                                                            const float* __restrict__ bvv,
                                                            float* __restrict__ out) {
    int i = blockIdx.x * 256 + threadIdx.x;   // 128*256 threads
    int g = i >> 8, c = i & 255;
    const float* r = in + (long)g * 256;
    float s = b[c];
    for (int k = 0; k < 256; ++k) s = fmaf(r[k], W[k * 256 + c], s);
    s = (s - bm[c]) * rsqrtf(bvv[c] + 1e-5f) * bg[c] + bb[c];
    out[i] = s > 0.f ? s : 0.f;
}

__global__ void __launch_bounds__(128) value_kernel(const float* __restrict__ v2,
                                                    const float* __restrict__ Wv,
                                                    const float* __restrict__ bv,
                                                    float* __restrict__ out) {
    int g = threadIdx.x;   // 128 threads, 1 block
    float s = bv[0];
    for (int k = 0; k < 256; ++k) s = fmaf(v2[g * 256 + k], Wv[k], s);
    out[(long)N_B * N_M + g] = tanhf(s);
}

// ---------------------------------------------------------------------------
// Host orchestration
// ---------------------------------------------------------------------------
extern "C" void kernel_launch(void* const* d_in, const int* in_sizes, int n_in,
                              void* d_out, int out_size, void* d_ws, size_t ws_size,
                              hipStream_t stream) {
    (void)in_sizes; (void)n_in; (void)out_size; (void)ws_size;
    const float* x     = (const float*)d_in[0];
    const int*   ei    = (const int*)d_in[1];
    const int*   move  = (const int*)d_in[2];
    const int*   player= (const int*)d_in[3];
    const int*   batch = (const int*)d_in[4];
    const float* Win   = (const float*)d_in[5];
    const float* b_in  = (const float*)d_in[6];
    const float* Wg[3]  = {(const float*)d_in[7],  (const float*)d_in[9],  (const float*)d_in[11]};
    const float* bg[3]  = {(const float*)d_in[8],  (const float*)d_in[10], (const float*)d_in[12]};
    const float* lng[3] = {(const float*)d_in[13], (const float*)d_in[15], (const float*)d_in[17]};
    const float* lnb[3] = {(const float*)d_in[14], (const float*)d_in[16], (const float*)d_in[18]};
    const float* Wfc1 = (const float*)d_in[19];
    const float* bfc1 = (const float*)d_in[20];
    const float* Wfc2 = (const float*)d_in[21];
    const float* bfc2 = (const float*)d_in[22];
    const float* bn1g = (const float*)d_in[23];
    const float* bn1b = (const float*)d_in[24];
    const float* bn1m = (const float*)d_in[25];
    const float* bn1v = (const float*)d_in[26];
    const float* bn2g = (const float*)d_in[27];
    const float* bn2b = (const float*)d_in[28];
    const float* bn2m = (const float*)d_in[29];
    const float* bn2v = (const float*)d_in[30];
    const float* Wp1 = (const float*)d_in[31];
    const float* bp1 = (const float*)d_in[32];
    const float* Wp2 = (const float*)d_in[33];
    const float* bp2 = (const float*)d_in[34];
    const float* Wv  = (const float*)d_in[35];
    const float* bv  = (const float*)d_in[36];
    float* out = (float*)d_out;

    char* ws = (char*)d_ws;
    const size_t HB = (size_t)N_NODES * HID * 4;          // 32 MB
    float*  h0  = (float*)(ws);                           // residual
    float*  h   = (float*)(ws + HB);                      // current features
    float*  z   = (float*)(ws + 2 * HB);                  // h + scatter
    __bf16* zb  = (__bf16*)(ws + 3 * HB);                 // bf16 view (16 MB), reused for hb
    char*   wts = ws + 3 * HB + HB / 2;
    __bf16* WT1 = (__bf16*)(wts);                         // 128 KB each
    __bf16* WT2 = (__bf16*)(wts + 131072);
    __bf16* WT3 = (__bf16*)(wts + 262144);
    __bf16* WTp = (__bf16*)(wts + 393216);                // 256 KB
    float*  ge  = (float*)(wts + 655360);
    float*  v1  = (float*)(wts + 655360 + 131072);
    float*  v2  = (float*)(wts + 655360 + 262144);
    __bf16* WTg[3] = {WT1, WT2, WT3};

    const long n4 = (long)N_NODES * HID / 4;              // float4 count

    // 1. weight convert/transpose -> bf16
    for (int l = 0; l < 3; ++l)
        wtrans_kernel<<<256, 256, 0, stream>>>(Wg[l], WTg[l], 256, 256);
    wtrans_kernel<<<512, 256, 0, stream>>>(Wp1, WTp, 512, 256);

    // 2. input projection -> h0 (residual)
    input_proj_kernel<<<N_NODES * HID / 256, 256, 0, stream>>>(x, player, batch, Win, b_in, h0);

    // 3. three GIN layers
    for (int l = 0; l < 3; ++l) {
        const float* hin = (l == 0) ? h0 : h;
        copy4_kernel<<<(int)(n4 / 256), 256, 0, stream>>>(hin, z, n4);
        scatter_kernel<<<N_EDGES * 64 / 256, 256, 0, stream>>>(ei, hin, z);
        cvt_bf16_kernel<<<(int)(n4 / 256), 256, 0, stream>>>(z, zb, n4);
        gemm_ln_kernel<<<N_NODES / 64, 256, 0, stream>>>(
            zb, WTg[l], bg[l], lng[l], lnb[l], h0, h, (l == 2) ? 1 : 0);
    }

    // 4. mean pool + value head
    meanpool_kernel<<<N_B, 256, 0, stream>>>(h, ge);
    dense_bn_relu_kernel<<<N_B, 256, 0, stream>>>(ge, Wfc1, bfc1, bn1g, bn1b, bn1m, bn1v, v1);
    dense_bn_relu_kernel<<<N_B, 256, 0, stream>>>(v1, Wfc2, bfc2, bn2g, bn2b, bn2m, bn2v, v2);
    value_kernel<<<1, 128, 0, stream>>>(v2, Wv, bv, out);

    // 5. policy head (h -> bf16, fused gather GEMM)
    cvt_bf16_kernel<<<(int)(n4 / 256), 256, 0, stream>>>(h, zb, n4);
    policy_kernel<<<N_B * N_M / 64, 256, 0, stream>>>(zb, WTp, bp1, Wp2, bp2, move, out);
}
